// MHA_Eager_90572270338233
// MI455X (gfx1250) — compile-verified, matrix operand order checked by reading, not run on hardware
//
#include <hip/hip_runtime.h>
#include <hip/hip_bf16.h>

#define B_  4
#define T_  2048
#define C_  1024
#define H_  16
#define HS  64

typedef __attribute__((ext_vector_type(16))) __bf16 v16bf;
typedef __attribute__((ext_vector_type(8)))  __bf16 v8bf;
typedef __attribute__((ext_vector_type(8)))  float  v8f;
typedef __attribute__((ext_vector_type(4)))  float  v4f;

static __device__ __forceinline__ __bf16 f2bf(float f) { return (__bf16)f; }

static __device__ __forceinline__ v8bf ld8(const __bf16* p) { return *(const v8bf*)p; }

static __device__ __forceinline__ v16bf mk16(v8bf lo, v8bf hi) {
    return __builtin_shufflevector(lo, hi, 0, 1, 2, 3, 4, 5, 6, 7,
                                           8, 9, 10, 11, 12, 13, 14, 15);
}

static __device__ __forceinline__ v8f wmma_bf16(v16bf a, v16bf b, v8f c) {
    return __builtin_amdgcn_wmma_f32_16x16x32_bf16(
        /*neg_a=*/false, a, /*neg_b=*/false, b,
        /*c_mod=*/(short)0, c, /*reuse_a=*/false, /*reuse_b=*/false);
}

// ---------------------------------------------------------------------------
// Kernel 0: one-time fp32 -> bf16 conversion (packed v_cvt_pk_bf16_f32),
// 8 elements / thread, b128 in -> b128 out.
// ---------------------------------------------------------------------------
__global__ __launch_bounds__(256) void cvt_f32_bf16(
    const float* __restrict__ in, __bf16* __restrict__ out, int n8)
{
    const int i = blockIdx.x * 256 + threadIdx.x;
    if (i < n8) {
        const v4f x0 = *(const v4f*)(in + (size_t)i * 8);
        const v4f x1 = *(const v4f*)(in + (size_t)i * 8 + 4);
        v8f x = __builtin_shufflevector(x0, x1, 0, 1, 2, 3, 4, 5, 6, 7);
        *(v8bf*)(out + (size_t)i * 8) = __builtin_convertvector(x, v8bf);
    }
}

// ---------------------------------------------------------------------------
// Kernel 1: qkv = x @ w_qkv^T  (M=8192, N=3072, K=1024), bf16 operands,
// 32x64 tile per wave: 2 A-frags x 4 B-frags -> 8 WMMAs per k-step.
// Q,K stored [B,H,T,64] bf16 ; V stored transposed [B,H,64,T] bf16.
// ---------------------------------------------------------------------------
__global__ __launch_bounds__(256) void mha_qkv_gemm(
    const __bf16* __restrict__ Xb, const __bf16* __restrict__ Wq,
    __bf16* __restrict__ Qb, __bf16* __restrict__ Kb, __bf16* __restrict__ Vt)
{
    const int lane = threadIdx.x & 31;
    const int lrow = lane & 15;
    const int hi   = lane >> 4;
    const int wave = blockIdx.x * 8 + (threadIdx.x >> 5);
    const int NT   = (3 * C_) / 64;               // 48 col tiles of 64
    const int mt   = wave / NT;
    const int nt   = wave % NT;

    const __bf16* a0 = Xb + (size_t)(mt * 32 + lrow) * C_;
    const __bf16* a1 = a0 + (size_t)16 * C_;
    const __bf16* b0 = Wq + (size_t)(nt * 64 + lrow) * C_;

    v8f acc[2][4] = {};
    for (int kk = 0; kk < C_; kk += 32) {
        const int o = kk + hi * 8;
        v16bf A0 = mk16(ld8(a0 + o), ld8(a0 + o + 16));
        v16bf A1 = mk16(ld8(a1 + o), ld8(a1 + o + 16));
#pragma unroll
        for (int c = 0; c < 4; ++c) {
            const __bf16* br = b0 + (size_t)(c * 16) * C_;
            v16bf Bf = mk16(ld8(br + o), ld8(br + o + 16));
            acc[0][c] = wmma_bf16(A0, Bf, acc[0][c]);
            acc[1][c] = wmma_bf16(A1, Bf, acc[1][c]);
        }
    }

    const int which = nt >> 4;                    // 0=Q 1=K 2=V (wave-uniform)
    const int h2    = nt & 15;                    // head (wave-uniform)
#pragma unroll
    for (int sub = 0; sub < 2; ++sub) {
        const int m0 = mt * 32 + sub * 16;
        const int bb = m0 >> 11;                  // batch
        const int bh = bb * H_ + h2;
        const int t0 = (m0 & (T_ - 1)) + hi * 8;  // 8 consecutive t per lane
        if (which == 2) {
#pragma unroll
            for (int c = 0; c < 4; ++c) {
                v8bf vv = __builtin_convertvector(acc[sub][c], v8bf);
                *(v8bf*)(Vt + ((size_t)bh * HS + c * 16 + lrow) * T_ + t0) = vv;
            }
        } else {
            __bf16* base = (which == 0 ? Qb : Kb) + (size_t)bh * T_ * HS;
#pragma unroll
            for (int c = 0; c < 4; ++c) {
                v8bf vv = __builtin_convertvector(acc[sub][c], v8bf);
                const int d = c * 16 + lrow;
#pragma unroll
                for (int r = 0; r < 8; ++r)
                    base[(size_t)(t0 + r) * HS + d] = vv[r];
            }
        }
    }
}

// ---------------------------------------------------------------------------
// Kernel 2: flash-style causal attention. One wave = one (b,h, 16-row q tile).
// ---------------------------------------------------------------------------
__global__ __launch_bounds__(256) void mha_flash_attn(
    const __bf16* __restrict__ Qb, const __bf16* __restrict__ Kb,
    const __bf16* __restrict__ Vt, __bf16* __restrict__ O)
{
    __shared__ __bf16 lds[8][16][32];             // per-wave P transpose staging
    const int lane = threadIdx.x & 31;
    const int lrow = lane & 15;
    const int hi   = lane >> 4;
    const int w    = threadIdx.x >> 5;
    const int wave = blockIdx.x * 8 + w;
    const int bh   = wave >> 7;                   // / (T/16)
    const int qt   = wave & 127;
    const int q0   = qt * 16;

    const __bf16* Q = Qb + (size_t)bh * T_ * HS;
    const __bf16* K = Kb + (size_t)bh * T_ * HS;
    const __bf16* V = Vt + (size_t)bh * HS * T_;

    // persistent Q A-fragments (hs=64 -> two K=32 chunks)
    const __bf16* qr = Q + (size_t)(q0 + lrow) * HS;
    v16bf qa0 = mk16(ld8(qr + hi * 8),      ld8(qr + hi * 8 + 16));
    v16bf qa1 = mk16(ld8(qr + 32 + hi * 8), ld8(qr + 32 + hi * 8 + 16));

    float mrun[8], lrun[8];
    v8f acc[4] = {};
#pragma unroll
    for (int r = 0; r < 8; ++r) { mrun[r] = -1e30f; lrun[r] = 0.f; }

    for (int j0 = 0; j0 < q0 + 16; j0 += 32) {
        // --- S = Q K^T for 32 keys (two 16x16 tiles, 4 WMMAs) ---
        const __bf16* k0 = K + (size_t)(j0 + lrow) * HS;
        const __bf16* k1 = K + (size_t)(j0 + 16 + lrow) * HS;
        v16bf kb00 = mk16(ld8(k0 + hi * 8),      ld8(k0 + hi * 8 + 16));
        v16bf kb01 = mk16(ld8(k0 + 32 + hi * 8), ld8(k0 + 32 + hi * 8 + 16));
        v16bf kb10 = mk16(ld8(k1 + hi * 8),      ld8(k1 + hi * 8 + 16));
        v16bf kb11 = mk16(ld8(k1 + 32 + hi * 8), ld8(k1 + 32 + hi * 8 + 16));

        v8f s0 = {}, s1 = {};
        s0 = wmma_bf16(qa0, kb00, s0); s0 = wmma_bf16(qa1, kb01, s0);
        s1 = wmma_bf16(qa0, kb10, s1); s1 = wmma_bf16(qa1, kb11, s1);

        // --- scale + causal mask + online softmax ---
        const int key0 = j0 + lrow, key1 = key0 + 16;
        float p0[8], p1[8], mx[8];
#pragma unroll
        for (int r = 0; r < 8; ++r) {
            const int qrow = q0 + r + hi * 8;
            p0[r] = (key0 <= qrow) ? s0[r] * 0.125f : -1e30f;
            p1[r] = (key1 <= qrow) ? s1[r] * 0.125f : -1e30f;
            mx[r] = fmaxf(p0[r], p1[r]);
        }
#pragma unroll
        for (int off = 1; off < 16; off <<= 1)
#pragma unroll
            for (int r = 0; r < 8; ++r)
                mx[r] = fmaxf(mx[r], __shfl_xor(mx[r], off, 32));

        float alpha[8];
#pragma unroll
        for (int r = 0; r < 8; ++r) {
            const float mn = fmaxf(mrun[r], mx[r]);
            alpha[r] = __expf(mrun[r] - mn);
            mrun[r]  = mn;
            p0[r] = __expf(p0[r] - mn);
            p1[r] = __expf(p1[r] - mn);
        }
        float rs[8];
#pragma unroll
        for (int r = 0; r < 8; ++r) rs[r] = p0[r] + p1[r];
#pragma unroll
        for (int off = 1; off < 16; off <<= 1)
#pragma unroll
            for (int r = 0; r < 8; ++r) rs[r] += __shfl_xor(rs[r], off, 32);
#pragma unroll
        for (int r = 0; r < 8; ++r) lrun[r] = lrun[r] * alpha[r] + rs[r];
#pragma unroll
        for (int c = 0; c < 4; ++c)
#pragma unroll
            for (int r = 0; r < 8; ++r) acc[c][r] *= alpha[r];

        // --- P: C-layout -> A-layout transpose through per-wave LDS slice ---
#pragma unroll
        for (int r = 0; r < 8; ++r) {
            lds[w][r + hi * 8][lrow]      = f2bf(p0[r]);
            lds[w][r + hi * 8][16 + lrow] = f2bf(p1[r]);
        }
        asm volatile("s_wait_dscnt 0" ::: "memory");  // per-wave LDS RAW fence
        v16bf pa = mk16(ld8(&lds[w][lrow][hi * 8]),
                        ld8(&lds[w][lrow][16 + hi * 8]));

        // --- O += P V (4 WMMAs; V^T layout -> contiguous B-fragments) ---
#pragma unroll
        for (int c = 0; c < 4; ++c) {
            const __bf16* vp = V + (size_t)(c * 16 + lrow) * T_ + j0;
            v16bf vb = mk16(ld8(vp + hi * 8), ld8(vp + 16 + hi * 8));
            acc[c] = wmma_bf16(pa, vb, acc[c]);
        }
    }

    // normalize and store to [B,T,C] bf16 (head-major within row)
    const int b = bh >> 4, h = bh & 15;
#pragma unroll
    for (int r = 0; r < 8; ++r) {
        const float inv = 1.0f / lrun[r];
        const int t = q0 + r + hi * 8;
        __bf16* orow = O + (size_t)(b * T_ + t) * C_ + h * HS;
#pragma unroll
        for (int c = 0; c < 4; ++c)
            orow[c * 16 + lrow] = f2bf(acc[c][r] * inv);
    }
}

// ---------------------------------------------------------------------------
// Kernel 3: out = attn @ w_proj^T + b_proj  (M=8192, N=1024, K=1024),
// bf16 operands, 32x64 tile per wave, f32 output.
// ---------------------------------------------------------------------------
__global__ __launch_bounds__(256) void mha_proj_gemm(
    const __bf16* __restrict__ A, const __bf16* __restrict__ Wp,
    const float* __restrict__ bias, float* __restrict__ out)
{
    const int lane = threadIdx.x & 31;
    const int lrow = lane & 15;
    const int hi   = lane >> 4;
    const int wave = blockIdx.x * 8 + (threadIdx.x >> 5);
    const int NT   = C_ / 64;                     // 16 col tiles of 64
    const int mt   = wave / NT;
    const int nt   = wave % NT;

    const __bf16* a0 = A  + (size_t)(mt * 32 + lrow) * C_;
    const __bf16* a1 = a0 + (size_t)16 * C_;
    const __bf16* b0 = Wp + (size_t)(nt * 64 + lrow) * C_;

    v8f acc[2][4] = {};
    for (int kk = 0; kk < C_; kk += 32) {
        const int o = kk + hi * 8;
        v16bf A0 = mk16(ld8(a0 + o), ld8(a0 + o + 16));
        v16bf A1 = mk16(ld8(a1 + o), ld8(a1 + o + 16));
#pragma unroll
        for (int c = 0; c < 4; ++c) {
            const __bf16* br = b0 + (size_t)(c * 16) * C_;
            v16bf Bf = mk16(ld8(br + o), ld8(br + o + 16));
            acc[0][c] = wmma_bf16(A0, Bf, acc[0][c]);
            acc[1][c] = wmma_bf16(A1, Bf, acc[1][c]);
        }
    }

#pragma unroll
    for (int c = 0; c < 4; ++c) {
        const int g = nt * 64 + c * 16 + lrow;
        const float bv = bias[g];
#pragma unroll
        for (int sub = 0; sub < 2; ++sub)
#pragma unroll
            for (int r = 0; r < 8; ++r) {
                const int m = mt * 32 + sub * 16 + r + hi * 8;
                out[(size_t)m * C_ + g] = acc[sub][c][r] + bv;
            }
    }
}

extern "C" void kernel_launch(void* const* d_in, const int* in_sizes, int n_in,
                              void* d_out, int out_size, void* d_ws, size_t ws_size,
                              hipStream_t stream) {
    const float* x     = (const float*)d_in[0];
    const float* wqkv  = (const float*)d_in[1];
    const float* wproj = (const float*)d_in[2];
    const float* bproj = (const float*)d_in[3];
    float* out = (float*)d_out;

    const size_t qkv_elems = (size_t)B_ * H_ * T_ * HS;   // 8,388,608
    const size_t x_elems   = (size_t)B_ * T_ * C_;        // 8,388,608
    const size_t wq_elems  = (size_t)3 * C_ * C_;         // 3,145,728
    const size_t wp_elems  = (size_t)C_ * C_;             // 1,048,576

    __bf16* Qb  = (__bf16*)d_ws;
    __bf16* Kb  = Qb  + qkv_elems;
    __bf16* Vt  = Kb  + qkv_elems;
    __bf16* O   = Vt  + qkv_elems;
    __bf16* Xb  = O   + qkv_elems;
    __bf16* Wqb = Xb  + x_elems;
    __bf16* Wpb = Wqb + wq_elems;                         // ~92 MB total

    // one-time conversions (x read once at 32MB; weights converted once
    // instead of per-M-tile -> GEMM loops are pure load_b128 + wmma)
    cvt_f32_bf16<<<(int)(x_elems / 8 / 256),  256, 0, stream>>>(x,     Xb,  (int)(x_elems / 8));
    cvt_f32_bf16<<<(int)(wq_elems / 8 / 256), 256, 0, stream>>>(wqkv,  Wqb, (int)(wq_elems / 8));
    cvt_f32_bf16<<<(int)(wp_elems / 8 / 256), 256, 0, stream>>>(wproj, Wpb, (int)(wp_elems / 8));

    mha_qkv_gemm<<<(256 * 48) / 8, 256, 0, stream>>>(Xb, Wqb, Qb, Kb, Vt);
    mha_flash_attn<<<(64 * 128) / 8, 256, 0, stream>>>(Qb, Kb, Vt, O);
    mha_proj_gemm<<<(256 * 16) / 8, 256, 0, stream>>>(O, Wpb, bproj, out);
}
